// ModuleCorrelationOptimized_44547400794797
// MI455X (gfx1250) — matrix operand order repaired and checked by reading, not compiled, not run on hardware
//
#include <hip/hip_runtime.h>

typedef __attribute__((ext_vector_type(16))) _Float16 v16h;
typedef __attribute__((ext_vector_type(8)))  float    v8f;

#define BB   4
#define CC   64
#define HH   128
#define WW   256
#define PAD  4
#define ND   9      // displacements per axis
#define TCH  81     // output channels
#define TILE 16

// One wave handles one (b, h, 16-wide w tile): computes a 16x32 Gram block
//   G[m, j] = sum_c first[c, w0+m] * second[c, h+dy, w0-4+j]
// with v_wmma_f32_16x16x32_f16 (K=64 in 2 chunks), then extracts the 16x9
// band out[m, dx] = G[m, m+dx+4] / 64 via LDS.
__global__ __launch_bounds__(256) void corr_wmma_f16(
    const float* __restrict__ first,
    const float* __restrict__ second,
    float* __restrict__ out) {
  const int lane = threadIdx.x & 31;
  const int wave = threadIdx.x >> 5;
  const int half = lane >> 4;   // lane half selects K sub-block per ISA layout
  const int l16  = lane & 15;

  const int tile = blockIdx.x * 8 + wave;   // 0..8191
  const int b    = tile >> 11;              // / (H * W/16) = /2048
  const int rem  = tile & 2047;
  const int h    = rem >> 4;
  const int w0   = (rem & 15) << 4;

  __shared__ float lds[8][TILE][34];        // per-wave 16x32 Gram block (+pad)

  const size_t HW = (size_t)HH * WW;

  // ---- A fragments (16x32 f16 per k-chunk), loaded once, reused for all dy.
  // ISA A layout: lane -> M=l16; VGPR v (elements e=2v, 2v+1):
  //   K = 2*(v&3) + 16*(v>>2) + 8*half + (e&1)
  // Lane-dependent parts (M=l16, 8*half channel block) folded into the base
  // pointer; the remaining per-element channel offset is a uniform immediate.
  const float* fptr = first + ((size_t)b * CC + (half << 3)) * HW
                            + (size_t)h * WW + w0 + l16;
  v16h afrag[2];
#pragma unroll
  for (int kc = 0; kc < 2; ++kc) {
    float ta[16];
#pragma unroll
    for (int e = 0; e < 16; ++e) {
      const int v = e >> 1, bit = e & 1;
      const int koff = ((v & 3) << 1) + ((v >> 2) << 4) + bit + (kc << 5);
      ta[e] = fptr[(size_t)koff * HW];      // immediate offset <= 7.2MB < 2^23
    }
#pragma unroll
    for (int e = 0; e < 16; ++e) afrag[kc][e] = (_Float16)ta[e];
  }

  // ---- Branch-free w-boundary handling: clamp address, multiply by 0/1 mask.
  const int wj0 = w0 - PAD + l16;           // Gram columns j = 0..15
  const int wj1 = w0 + 12 + l16;            // Gram columns j = 16..31 (>23 wasted)
  const int wj0c = wj0 < 0 ? 0 : wj0;       // wj0 <= 251, only low clamp needed
  const int wj1c = wj1 > (WW - 1) ? (WW - 1) : wj1;  // wj1 >= 12, only high clamp
  const float m0 = (wj0 >= 0) ? 1.0f : 0.0f;
  const float m1 = (wj1 < WW) ? 1.0f : 0.0f;

  const float* sbase = second + ((size_t)b * CC + (half << 3)) * HW;

  for (int d = 0; d < ND; ++d) {            // d = dy + 4
    const int hh = h + d - PAD;             // wave-uniform -> EXEC stays all-1s
    v8f acc0 = {};
    v8f acc1 = {};
    if (hh >= 0 && hh < HH) {
      const float* p0 = sbase + (size_t)hh * WW + wj0c;
      const float* p1 = sbase + (size_t)hh * WW + wj1c;
      if (d + 1 < ND && hh + 1 < HH)
        __builtin_prefetch(p0 + WW, 0, 3);  // global_prefetch_b8: next dy row
#pragma unroll
      for (int kc = 0; kc < 2; ++kc) {
        // ISA B layout (32x16 f16): lane -> N=l16; element e -> K = 8*half + (e<8 ? e : e+8)
        float t0[16], t1[16];
#pragma unroll
        for (int e = 0; e < 16; ++e) {
          const size_t koff = (size_t)((kc << 5) + (e < 8 ? e : e + 8)) * HW;  // uniform immediate
          t0[e] = p0[koff];
          t1[e] = p1[koff];
        }
        v16h b0, b1;
#pragma unroll
        for (int e = 0; e < 16; ++e) {
          b0[e] = (_Float16)(t0[e] * m0);
          b1[e] = (_Float16)(t1[e] * m1);
        }
        acc0 = __builtin_amdgcn_wmma_f32_16x16x32_f16(
            false, afrag[kc], false, b0, (short)0, acc0, false, false);
        acc1 = __builtin_amdgcn_wmma_f32_16x16x32_f16(
            false, afrag[kc], false, b1, (short)0, acc1, false, false);
      }
    }

    // ---- Spill Gram block to LDS. C/D layout: VGPR r -> row M = r + 8*half,
    // lane -> col N = l16. Same-wave DS ops are in-order (DScnt), so the
    // subsequent re-reads are safe without a workgroup barrier.
#pragma unroll
    for (int r = 0; r < 8; ++r) {
      lds[wave][r + (half << 3)][l16]      = acc0[r];
      lds[wave][r + (half << 3)][16 + l16] = acc1[r];
    }

    // ---- Band extraction + coalesced store: out[m, dx] = G[m][m + dx] / C
#pragma unroll
    for (int i = 0; i < 5; ++i) {
      const int t = i * 32 + lane;          // 144 = 16 pixels * 9 dx values
      if (t < 144) {
        const int m   = t & 15;
        const int dxi = t >> 4;             // 0..8
        const float val = lds[wave][m][m + dxi] * (1.0f / (float)CC);
        const int tc = d * ND + dxi;        // (dy+4)*9 + (dx+4)
        out[(((size_t)b * TCH + tc) * HH + h) * WW + w0 + m] = val;
      }
    }
  }
}

extern "C" void kernel_launch(void* const* d_in, const int* in_sizes, int n_in,
                              void* d_out, int out_size, void* d_ws, size_t ws_size,
                              hipStream_t stream) {
  const float* first  = (const float*)d_in[0];
  const float* second = (const float*)d_in[1];
  float* out = (float*)d_out;
  // 8192 wave-tiles, 8 waves (256 threads) per block
  corr_wmma_f16<<<dim3(1024), dim3(256), 0, stream>>>(first, second, out);
  (void)in_sizes; (void)n_in; (void)out_size; (void)d_ws; (void)ws_size;
}
